// GPT2Block_Attention_35691178230206
// MI455X (gfx1250) — matrix operand order repaired and checked by reading, not compiled, not run on hardware
//
#include <hip/hip_runtime.h>
#include <hip/hip_bf16.h>
#include <math.h>

typedef __bf16 bf16;
typedef bf16 v16bf __attribute__((ext_vector_type(16)));
typedef bf16 v8bf  __attribute__((ext_vector_type(8)));
typedef bf16 v4bf  __attribute__((ext_vector_type(4)));
typedef float v8f  __attribute__((ext_vector_type(8)));

typedef unsigned int u32x4 __attribute__((ext_vector_type(4)));
typedef int i32x4 __attribute__((ext_vector_type(4)));
typedef int i32x8 __attribute__((ext_vector_type(8)));

#define B_   4
#define S_   2048
#define H_   1024
#define NH_  16
#define HS_  64
#define ROWS_ (B_ * S_)   // 8192

// GEMM tiling
#define TM  128
#define TN  128
#define TK  64
#define LDT 72            // padded LDS row stride (bf16 elems, 144B)

// flash tiling
#define FQ  64            // query rows per workgroup
#define FK  128           // key block
#define LDK 72            // 64 data + 8 pad bf16 (= 32 + 4 dwords)
#define LDV 136           // 128 data + 8 pad bf16 (= 64 + 4 dwords)

__device__ __forceinline__ v8f zero8() {
  v8f z = {0.f, 0.f, 0.f, 0.f, 0.f, 0.f, 0.f, 0.f};
  return z;
}

__device__ __forceinline__ v8f wmma_bf16f32(v16bf a, v16bf b, v8f c) {
  return __builtin_amdgcn_wmma_f32_16x16x32_bf16(false, a, false, b, (short)0, c,
                                                 false, false);
}

// ---- CDNA5 async global->LDS (bypasses VGPRs, tracked on ASYNCcnt) --------
__device__ __forceinline__ void async_copy_b128(void* lptr, const void* gptr) {
  unsigned lds = (unsigned)(uintptr_t)lptr;              // flat[31:0] = LDS addr
  unsigned long long ga = (unsigned long long)(uintptr_t)gptr;
  asm volatile("global_load_async_to_lds_b128 %0, %1, off"
               :: "v"(lds), "v"(ga) : "memory");
}
__device__ __forceinline__ void wait_async0() {
  asm volatile("s_wait_asynccnt 0x0" ::: "memory");
}

// ---- CDNA5 TDM: 2D tile load with LDS padding (ISA 08_async_tensor §8) ----
// dims/strides in dword (4-byte) units; pad codes per D# group-1 encoding.
__device__ __forceinline__ void tdm_load_2d(void* lptr, const void* gptr,
                                            unsigned tensor_d0, unsigned tensor_d1,
                                            unsigned tile_d0, unsigned tile_d1,
                                            unsigned long long d0_stride,
                                            unsigned pad_interval_code,
                                            unsigned pad_amount_code)
{
  unsigned long long ga = (unsigned long long)(uintptr_t)gptr;
  unsigned lds = (unsigned)(uintptr_t)lptr;
  u32x4 g0;
  g0[0] = 1u;                                          // count=1 (valid), user mode
  g0[1] = lds;                                         // lds_addr [63:32]
  g0[2] = (unsigned)(ga & 0xffffffffu);                // global_addr [95:64]
  g0[3] = (unsigned)((ga >> 32) & 0x01ffffffu)         // global_addr [120:96]
        | (2u << 30);                                  // type=2 ("image")
  i32x8 g1;
  unsigned w0 = (2u << 16)                             // data_size = 4 bytes
              | (1u << 20)                             // pad_enable
              | (pad_interval_code << 22)
              | (pad_amount_code << 25);
  g1[0] = (int)w0;
  g1[1] = (int)((tensor_d0 & 0xffffu) << 16);          // tensor_dim0[15:0] @ [63:48]
  g1[2] = (int)(((tensor_d0 >> 16) & 0xffffu) |
                ((tensor_d1 & 0xffffu) << 16));        // dim0[31:16] | dim1[15:0]
  g1[3] = (int)(((tensor_d1 >> 16) & 0xffffu) |
                ((tile_d0 & 0xffffu) << 16));          // dim1[31:16] | tile_dim0
  g1[4] = (int)(tile_d1 & 0xffffu);                    // tile_dim1 (tile_dim2=0)
  g1[5] = (int)(unsigned)(d0_stride & 0xffffffffull);  // dim0_stride[31:0]
  g1[6] = (int)(unsigned)((d0_stride >> 32) & 0xffffull); // dim0_stride[47:32]
  g1[7] = 0;
  i32x4 z4 = {0, 0, 0, 0};
#if __clang_major__ >= 23
  i32x8 z8 = {0, 0, 0, 0, 0, 0, 0, 0};
  __builtin_amdgcn_tensor_load_to_lds(g0, g1, z4, z4, z8, 0);
#else
  __builtin_amdgcn_tensor_load_to_lds(g0, g1, z4, z4, 0);
#endif
}

// A fragment 16x32 bf16: lanes 0-15: M=lane, K=0..7 & 16..23;
// lanes 16-31: M=lane-16, K=8..15 & 24..31
__device__ __forceinline__ v16bf load_a_frag(const bf16* tile, int ld, int lane) {
  int row  = lane & 15;
  int koff = (lane >> 4) * 8;
  union { v16bf v; v8bf h[2]; } u;
  const bf16* p = tile + row * ld + koff;
  u.h[0] = *(const v8bf*)(p);
  u.h[1] = *(const v8bf*)(p + 16);
  return u.v;
}

// B fragment 32x16 bf16 from [N][K] storage: lane n / n+16 hold column n's
// K=0..7 & 16..23 / 8..15 & 24..31 (same order as A).
__device__ __forceinline__ v16bf load_b_frag(const bf16* tile, int ld, int lane) {
  int n    = lane & 15;
  int koff = (lane >> 4) * 8;
  union { v16bf v; v8bf h[2]; } u;
  const bf16* p = tile + n * ld + koff;
  u.h[0] = *(const v8bf*)(p);
  u.h[1] = *(const v8bf*)(p + 16);
  return u.v;
}

// ---------------------------------------------------------------------------
// Kernel 1: fused Q/K/V projection (z selects W). q,k -> [B,NH,S,HS];
// v -> [B,NH,HS,S] (transposed for PV B-fragments).
// ---------------------------------------------------------------------------
__global__ __launch_bounds__(256)
void qkv_proj_kernel(const float* __restrict__ x,
                     const float* __restrict__ Wq, const float* __restrict__ bq,
                     const float* __restrict__ Wk, const float* __restrict__ bk,
                     const float* __restrict__ Wv, const float* __restrict__ bv,
                     bf16* __restrict__ qo, bf16* __restrict__ ko,
                     bf16* __restrict__ vo)
{
  __shared__ __align__(16) bf16 sA[TM * LDT];
  __shared__ __align__(16) bf16 sB[TN * LDT];

  const int tid  = threadIdx.x;
  const int lane = tid & 31;
  const int wave = tid >> 5;
  const int wm   = wave >> 1;
  const int wn   = wave & 1;
  const int n0   = blockIdx.x * TN;
  const int m0   = blockIdx.y * TM;
  const int z    = blockIdx.z;

  const float* W    = (z == 0) ? Wq : (z == 1) ? Wk : Wv;
  const float* bias = (z == 0) ? bq : (z == 1) ? bk : bv;

  v8f acc[2][4];
  for (int i = 0; i < 2; ++i)
    for (int j = 0; j < 4; ++j)
      acc[i][j] = zero8();

  for (int kt = 0; kt < H_ / TK; ++kt) {
    // stage x and W slabs (f32 -> bf16) through VGPRs (conversion required)
    #pragma unroll
    for (int i = 0; i < 8; ++i) {
      int f   = tid + i * 256;
      int row = f >> 4;
      int c4  = f & 15;
      float4 a = *(const float4*)(x + (size_t)(m0 + row) * H_ + kt * TK + c4 * 4);
      v4bf pa = {(bf16)a.x, (bf16)a.y, (bf16)a.z, (bf16)a.w};
      *(v4bf*)(&sA[row * LDT + c4 * 4]) = pa;
      float4 b = *(const float4*)(W + (size_t)(n0 + row) * H_ + kt * TK + c4 * 4);
      v4bf pb = {(bf16)b.x, (bf16)b.y, (bf16)b.z, (bf16)b.w};
      *(v4bf*)(&sB[row * LDT + c4 * 4]) = pb;
    }
    if (kt + 1 < H_ / TK) {
      __builtin_prefetch(x + (size_t)(m0 + (tid >> 1)) * H_ + (kt + 1) * TK, 0, 1);
      __builtin_prefetch(W + (size_t)(n0 + (tid >> 1)) * H_ + (kt + 1) * TK, 0, 1);
    }
    __syncthreads();

    #pragma unroll
    for (int ks = 0; ks < 2; ++ks) {
      v16bf afr[2], bfr[4];
      #pragma unroll
      for (int i = 0; i < 2; ++i)
        afr[i] = load_a_frag(&sA[(wm * 32 + i * 16) * LDT + ks * 32], LDT, lane);
      #pragma unroll
      for (int j = 0; j < 4; ++j)
        bfr[j] = load_b_frag(&sB[(wn * 64 + j * 16) * LDT + ks * 32], LDT, lane);
      #pragma unroll
      for (int i = 0; i < 2; ++i)
        #pragma unroll
        for (int j = 0; j < 4; ++j)
          acc[i][j] = wmma_bf16f32(afr[i], bfr[j], acc[i][j]);
    }
    __syncthreads();
  }

  const int mhalf = lane >> 4;
  #pragma unroll
  for (int i = 0; i < 2; ++i) {
    #pragma unroll
    for (int j = 0; j < 4; ++j) {
      int n = n0 + wn * 64 + j * 16 + (lane & 15);
      float bvv = bias[n];
      int h = n >> 6, d = n & 63;
      #pragma unroll
      for (int r = 0; r < 8; ++r) {
        int m  = m0 + wm * 32 + i * 16 + r + mhalf * 8;
        int bb = m >> 11;
        int s  = m & (S_ - 1);
        float val = acc[i][j][r] + bvv;
        if (z == 2) {
          vo[((size_t)(bb * NH_ + h) * HS_ + d) * S_ + s] = (bf16)val;
        } else if (z == 0) {
          qo[((size_t)(bb * NH_ + h) * S_ + s) * HS_ + d] = (bf16)val;
        } else {
          ko[((size_t)(bb * NH_ + h) * S_ + s) * HS_ + d] = (bf16)val;
        }
      }
    }
  }
}

// ---------------------------------------------------------------------------
// Kernel 2: causal flash attention; 64 query rows per workgroup, 4 waves.
// Q staged via async-to-LDS; K / V^T tiles via TDM with hardware LDS padding.
// ---------------------------------------------------------------------------
__global__ __launch_bounds__(128)
void flash_attn_kernel(const bf16* __restrict__ q, const bf16* __restrict__ k,
                       const bf16* __restrict__ vt, bf16* __restrict__ attn)
{
  __shared__ __align__(16) bf16  sQ[FQ * LDK];
  __shared__ __align__(16) bf16  sK[FK * LDK];
  __shared__ __align__(16) bf16  sVT[HS_ * LDV];
  __shared__ __align__(16) float sS[FQ * FK];
  __shared__ __align__(16) bf16  sP[FQ * LDV];
  __shared__ float sScale[FQ];
  __shared__ float sSum[FQ];

  const int tid  = threadIdx.x;
  const int lane = tid & 31;
  const int wave = tid >> 5;
  const int qb   = blockIdx.x;
  const int bh   = blockIdx.y;
  const int b    = bh / NH_;
  const int h    = bh % NH_;
  const int q0   = qb * FQ;

  const size_t head = (size_t)bh * S_ * HS_;

  // stage Q (64x64 bf16) asynchronously: global -> LDS, no VGPR pass
  #pragma unroll
  for (int i = 0; i < 4; ++i) {
    int g = tid + i * 128;
    int row = g >> 3, c8 = g & 7;
    async_copy_b128(&sQ[row * LDK + c8 * 8],
                    q + head + (size_t)(q0 + row) * HS_ + c8 * 8);
  }

  v8f acco[4];
  for (int j = 0; j < 4; ++j) acco[j] = zero8();

  const int rrow = wave * 16 + (lane >> 1);
  const int half = lane & 1;
  float m_run = -__builtin_inff();
  float l_run = 0.0f;

  const int kb_max = (q0 + FQ - 1) >> 7;
  for (int kb = 0; kb <= kb_max; ++kb) {
    const int k0 = kb * FK;

    // TDM: K tile 128 rows x 32 dwords (stride 32), pad 4 dwords/row -> LDK=72
    if (wave == 0) {
      tdm_load_2d(&sK[0], k + head + (size_t)k0 * HS_,
                  /*tensor_d0=*/32, /*tensor_d1=*/FK,
                  /*tile_d0=*/32, /*tile_d1=*/FK,
                  /*d0_stride=*/32ull,
                  /*pad_interval(32dw)=*/4u, /*pad_amount(4dw)=*/3u);
    }
    // TDM: V^T tile 64 rows x 64 dwords (row stride S/2 dwords) -> LDV=136
    if (wave == 1) {
      tdm_load_2d(&sVT[0], vt + head + k0,
                  /*tensor_d0=*/S_ / 2, /*tensor_d1=*/HS_,
                  /*tile_d0=*/64, /*tile_d1=*/HS_,
                  /*d0_stride=*/(unsigned long long)(S_ / 2),
                  /*pad_interval(64dw)=*/5u, /*pad_amount(4dw)=*/3u);
    }
    wait_async0();                         // Q copies (ASYNCcnt)
    if (wave < 2) __builtin_amdgcn_s_wait_tensorcnt(0);
    __syncthreads();

    // scores: 16 query rows x 128 keys, K-dim = HS (2 steps of 32)
    v8f sacc[8];
    #pragma unroll
    for (int j = 0; j < 8; ++j) sacc[j] = zero8();
    #pragma unroll
    for (int ks = 0; ks < 2; ++ks) {
      v16bf afr = load_a_frag(&sQ[(wave * 16) * LDK + ks * 32], LDK, lane);
      #pragma unroll
      for (int j = 0; j < 8; ++j) {
        v16bf bfr = load_b_frag(&sK[(j * 16) * LDK + ks * 32], LDK, lane);
        sacc[j] = wmma_bf16f32(afr, bfr, sacc[j]);
      }
    }
    {
      const int mhalf = lane >> 4;
      #pragma unroll
      for (int j = 0; j < 8; ++j) {
        int col = j * 16 + (lane & 15);
        int key = k0 + col;
        #pragma unroll
        for (int r = 0; r < 8; ++r) {
          int rowm = wave * 16 + r + mhalf * 8;
          int qry  = q0 + rowm;
          float sv = sacc[j][r] * 0.125f;   // 1/sqrt(64)
          sS[rowm * FK + col] = (key <= qry) ? sv : -3.0e38f;
        }
      }
    }
    __syncthreads();

    // online softmax: two lanes per row, 64 columns each
    {
      const float* srow = &sS[rrow * FK + half * 64];
      float mb = -3.0e38f;
      #pragma unroll 8
      for (int c = 0; c < 64; ++c) mb = fmaxf(mb, srow[c]);
      mb = fmaxf(mb, __shfl_xor(mb, 1, 32));
      float m_new = fmaxf(m_run, mb);
      float scl = __expf(m_run - m_new);
      float lb = 0.0f;
      bf16* prow = &sP[rrow * LDV + half * 64];
      #pragma unroll 8
      for (int c = 0; c < 64; ++c) {
        float p = __expf(srow[c] - m_new);
        prow[c] = (bf16)p;
        lb += p;
      }
      lb += __shfl_xor(lb, 1, 32);
      l_run = l_run * scl + lb;
      m_run = m_new;
      if (half == 0) sScale[rrow] = scl;
    }
    __syncthreads();

    // rescale accumulators, then O += P @ V (K-dim = 128 keys, 4 steps)
    {
      const int mhalf = lane >> 4;
      #pragma unroll
      for (int j = 0; j < 4; ++j) {
        #pragma unroll
        for (int r = 0; r < 8; ++r) {
          int rowm = wave * 16 + r + mhalf * 8;
          acco[j][r] *= sScale[rowm];
        }
      }
    }
    #pragma unroll
    for (int ks = 0; ks < 4; ++ks) {
      v16bf afr = load_a_frag(&sP[(wave * 16) * LDV + ks * 32], LDV, lane);
      #pragma unroll
      for (int j = 0; j < 4; ++j) {
        v16bf bfr = load_b_frag(&sVT[(j * 16) * LDV + ks * 32], LDV, lane);
        acco[j] = wmma_bf16f32(afr, bfr, acco[j]);
      }
    }
    __syncthreads();
  }

  if (half == 0) sSum[rrow] = l_run;
  __syncthreads();

  const int mhalf = lane >> 4;
  #pragma unroll
  for (int j = 0; j < 4; ++j) {
    int d = j * 16 + (lane & 15);
    #pragma unroll
    for (int r = 0; r < 8; ++r) {
      int rowm = wave * 16 + r + mhalf * 8;
      float inv = 1.0f / sSum[rowm];
      int sg = q0 + rowm;
      attn[((size_t)(b * S_ + sg)) * H_ + h * 64 + d] = (bf16)(acco[j][r] * inv);
    }
  }
}

// ---------------------------------------------------------------------------
// Kernel 3: output projection — attn(bf16) @ Wp^T + bp -> f32 d_out.
// A slab staged with async-to-LDS (already bf16); Wp converted via VGPRs.
// ---------------------------------------------------------------------------
__global__ __launch_bounds__(256)
void out_proj_kernel(const bf16* __restrict__ a, const float* __restrict__ Wp,
                     const float* __restrict__ bp, float* __restrict__ out)
{
  __shared__ __align__(16) bf16 sA[TM * LDT];
  __shared__ __align__(16) bf16 sB[TN * LDT];

  const int tid  = threadIdx.x;
  const int lane = tid & 31;
  const int wave = tid >> 5;
  const int wm   = wave >> 1;
  const int wn   = wave & 1;
  const int n0   = blockIdx.x * TN;
  const int m0   = blockIdx.y * TM;

  v8f acc[2][4];
  for (int i = 0; i < 2; ++i)
    for (int j = 0; j < 4; ++j)
      acc[i][j] = zero8();

  for (int kt = 0; kt < H_ / TK; ++kt) {
    #pragma unroll
    for (int i = 0; i < 4; ++i) {       // A slab: bf16, async global->LDS
      int g = tid + i * 256;
      int row = g >> 3, c8 = g & 7;
      async_copy_b128(&sA[row * LDT + c8 * 8],
                      a + (size_t)(m0 + row) * H_ + kt * TK + c8 * 8);
    }
    #pragma unroll
    for (int i = 0; i < 8; ++i) {       // B slab: Wp f32 -> bf16 via VGPRs
      int f   = tid + i * 256;
      int row = f >> 4;
      int c4  = f & 15;
      float4 w = *(const float4*)(Wp + (size_t)(n0 + row) * H_ + kt * TK + c4 * 4);
      v4bf pw = {(bf16)w.x, (bf16)w.y, (bf16)w.z, (bf16)w.w};
      *(v4bf*)(&sB[row * LDT + c4 * 4]) = pw;
    }
    wait_async0();
    __syncthreads();

    #pragma unroll
    for (int ks = 0; ks < 2; ++ks) {
      v16bf afr[2], bfr[4];
      #pragma unroll
      for (int i = 0; i < 2; ++i)
        afr[i] = load_a_frag(&sA[(wm * 32 + i * 16) * LDT + ks * 32], LDT, lane);
      #pragma unroll
      for (int j = 0; j < 4; ++j)
        bfr[j] = load_b_frag(&sB[(wn * 64 + j * 16) * LDT + ks * 32], LDT, lane);
      #pragma unroll
      for (int i = 0; i < 2; ++i)
        #pragma unroll
        for (int j = 0; j < 4; ++j)
          acc[i][j] = wmma_bf16f32(afr[i], bfr[j], acc[i][j]);
    }
    __syncthreads();
  }

  const int mhalf = lane >> 4;
  #pragma unroll
  for (int i = 0; i < 2; ++i) {
    #pragma unroll
    for (int j = 0; j < 4; ++j) {
      int n = n0 + wn * 64 + j * 16 + (lane & 15);
      float bvv = bp[n];
      #pragma unroll
      for (int r = 0; r < 8; ++r) {
        int m = m0 + wm * 32 + i * 16 + r + mhalf * 8;
        out[(size_t)m * H_ + n] = acc[i][j][r] + bvv;
      }
    }
  }
}

// ---------------------------------------------------------------------------
extern "C" void kernel_launch(void* const* d_in, const int* in_sizes, int n_in,
                              void* d_out, int out_size, void* d_ws, size_t ws_size,
                              hipStream_t stream) {
  const float* x  = (const float*)d_in[0];
  const float* Wq = (const float*)d_in[1];
  const float* bq = (const float*)d_in[2];
  const float* Wk = (const float*)d_in[3];
  const float* bk = (const float*)d_in[4];
  const float* Wv = (const float*)d_in[5];
  const float* bv = (const float*)d_in[6];
  const float* Wp = (const float*)d_in[7];
  const float* bp = (const float*)d_in[8];
  float* out = (float*)d_out;

  char* ws = (char*)d_ws;
  const size_t MB = 1024ull * 1024ull;
  bf16* qb = (bf16*)(ws + 0 * MB);    // [B,NH,S,HS] bf16 : 16 MB
  bf16* kb = (bf16*)(ws + 16 * MB);   // [B,NH,S,HS] bf16 : 16 MB
  bf16* vb = (bf16*)(ws + 32 * MB);   // [B,NH,HS,S] bf16 : 16 MB (transposed)
  bf16* ab = (bf16*)(ws + 48 * MB);   // [B,S,H]     bf16 : 16 MB

  (void)in_sizes; (void)n_in; (void)out_size; (void)ws_size;

  qkv_proj_kernel<<<dim3(H_ / TN, ROWS_ / TM, 3), 256, 0, stream>>>(
      x, Wq, bq, Wk, bk, Wv, bv, qb, kb, vb);
  flash_attn_kernel<<<dim3(S_ / FQ, B_ * NH_), 128, 0, stream>>>(qb, kb, vb, ab);
  out_proj_kernel<<<dim3(H_ / TN, ROWS_ / TM), 256, 0, stream>>>(ab, Wp, bp, out);
}
